// MOELayer_73134703116966
// MI455X (gfx1250) — matrix operand order repaired and checked by reading, not compiled, not run on hardware
//
#include <hip/hip_runtime.h>
#include <hip/hip_bf16.h>
#include <stdint.h>

// Problem constants (match reference)
#define B_TOK 16384
#define D_DIM 512
#define DFF   2048
#define E_NUM 8
#define MTILE 32   // tokens per expert tile

typedef __bf16 bf16_t;
typedef __attribute__((ext_vector_type(16))) __bf16 v16bf;
typedef __attribute__((ext_vector_type(8)))  __bf16 v8bf;
typedef __attribute__((ext_vector_type(8)))  float  v8f;

static __device__ __forceinline__ bf16_t f2bf(float f) {
  uint32_t u = __builtin_bit_cast(uint32_t, f);
  u += 0x7FFFu + ((u >> 16) & 1u);          // round-to-nearest-even to bf16
  uint16_t h = (uint16_t)(u >> 16);
  return __builtin_bit_cast(bf16_t, h);
}

static __device__ __forceinline__ v16bf cat16(v8bf lo, v8bf hi) {
  return __builtin_shufflevector(lo, hi, 0,1,2,3,4,5,6,7,8,9,10,11,12,13,14,15);
}

// ---------------------------------------------------------------------------
// Kernel 1: tiled transpose + fp32->bf16 convert.
// src: [Z][R][C] fp32  ->  dst: [Z][C][R] bf16
// block (32,8), grid (C/32, R/32, Z)
// ---------------------------------------------------------------------------
__global__ __launch_bounds__(256) void transpose_convert_kernel(
    const float* __restrict__ src, bf16_t* __restrict__ dst, int R, int C) {
  __shared__ float tile[32][33];
  const size_t zoff = (size_t)blockIdx.z * R * C;
  const float* s = src + zoff;
  bf16_t* d = dst + zoff;
  const int c0 = blockIdx.x * 32, r0 = blockIdx.y * 32;
  const int tx = threadIdx.x, ty = threadIdx.y;
  #pragma unroll
  for (int i = ty; i < 32; i += 8)
    tile[i][tx] = s[(size_t)(r0 + i) * C + (c0 + tx)];
  __syncthreads();
  #pragma unroll
  for (int i = ty; i < 32; i += 8)
    d[(size_t)(c0 + i) * R + (r0 + tx)] = f2bf(tile[tx][i]);
}

// ---------------------------------------------------------------------------
// Kernel 2: routing. One wave32 per token.
// Computes logits = x@Wg, softmax, top-2; appends token to per-expert lists;
// converts the token's x row to bf16; initializes out with routed b2 term.
// block 256 (8 waves -> 8 tokens), grid B/8
// ---------------------------------------------------------------------------
__global__ __launch_bounds__(256) void routing_kernel(
    const float* __restrict__ x, const float* __restrict__ Wg,
    const float* __restrict__ b2, bf16_t* __restrict__ xb,
    int* __restrict__ cnt, int* __restrict__ tokIdx, float* __restrict__ tokW,
    float* __restrict__ out) {
  const int lane = threadIdx.x & 31;
  const int wv   = threadIdx.x >> 5;
  const int b    = blockIdx.x * 8 + wv;

  float acc[E_NUM];
  #pragma unroll
  for (int e = 0; e < E_NUM; ++e) acc[e] = 0.f;

  const float* xrow = x + (size_t)b * D_DIM;
  bf16_t* xbrow = xb + (size_t)b * D_DIM;
  for (int d = lane; d < D_DIM; d += 32) {
    float xv = xrow[d];
    xbrow[d] = f2bf(xv);
    #pragma unroll
    for (int e = 0; e < E_NUM; ++e) acc[e] += xv * Wg[d * E_NUM + e];
  }
  #pragma unroll
  for (int e = 0; e < E_NUM; ++e) {
    #pragma unroll
    for (int off = 16; off >= 1; off >>= 1)
      acc[e] += __shfl_down(acc[e], off, 32);
  }

  int e0 = 0, e1 = 0;
  float v0 = 0.f, v1 = 0.f;
  if (lane == 0) {
    float m = acc[0];
    #pragma unroll
    for (int e = 1; e < E_NUM; ++e) m = fmaxf(m, acc[e]);
    float p[E_NUM], s = 0.f;
    #pragma unroll
    for (int e = 0; e < E_NUM; ++e) { p[e] = __expf(acc[e] - m); s += p[e]; }
    #pragma unroll
    for (int e = 1; e < E_NUM; ++e) if (p[e] > p[e0]) e0 = e;   // argmax (lowest idx on tie)
    e1 = (e0 == 0) ? 1 : 0;
    #pragma unroll
    for (int e = 0; e < E_NUM; ++e) if (e != e0 && p[e] > p[e1]) e1 = e;
    v0 = p[e0] / s; v1 = p[e1] / s;
    int p0 = atomicAdd(&cnt[e0], 1);
    tokIdx[e0 * B_TOK + p0] = b; tokW[e0 * B_TOK + p0] = v0;
    int p1 = atomicAdd(&cnt[e1], 1);
    tokIdx[e1 * B_TOK + p1] = b; tokW[e1 * B_TOK + p1] = v1;
  }
  e0 = __shfl(e0, 0, 32); e1 = __shfl(e1, 0, 32);
  v0 = __shfl(v0, 0, 32); v1 = __shfl(v1, 0, 32);

  const float* b2r0 = b2 + (size_t)e0 * D_DIM;
  const float* b2r1 = b2 + (size_t)e1 * D_DIM;
  float* orow = out + (size_t)b * D_DIM;
  for (int c = lane; c < D_DIM; c += 32)
    orow[c] = v0 * b2r0[c] + v1 * b2r1[c];
}

// ---------------------------------------------------------------------------
// Kernel 3: fused expert FFN over a 32-token tile of one expert.
// grid (B/MTILE, E), block 256 (8 waves). LDS: xg 33.3KB + h 131.6KB.
// Each wave register-blocks 2 M-tiles x 4 N-tiles: per k-step,
// 2 A fragments + 4 B fragments feed 8 WMMAs (1 global b128 load per WMMA).
// WMMA fragment layouts per CDNA5 ISA 7.12.2:
//  A (16x32 bf16): lane = h*16+m holds row m, K chunks [k0+h*8, +8) and [k0+16+h*8, +8)
//  B (32x16 bf16): lane = h*16+n holds col n, K chunks [k0+h*8, +8) and [k0+16+h*8, +8)
//  C/D (16x16 f32): lane = h*16+n holds col n, rows m = h*8 + i (i = vgpr index)
// ---------------------------------------------------------------------------
__global__ __launch_bounds__(256) void moe_ffn_kernel(
    const bf16_t* __restrict__ xb, const bf16_t* __restrict__ W1T,
    const float* __restrict__ b1, const bf16_t* __restrict__ W2T,
    const int* __restrict__ cnt, const int* __restrict__ tokIdx,
    const float* __restrict__ tokW, float* __restrict__ out) {
  __shared__ v8bf xg[MTILE][D_DIM / 8 + 1];   // 32 x 65 x 16B, +1 chunk pad vs banks
  __shared__ v8bf hb[MTILE][DFF / 8 + 1];     // 32 x 257 x 16B
  __shared__ int   stok[MTILE];
  __shared__ float sw[MTILE];

  const int e    = blockIdx.y;
  const int Ne   = cnt[e];
  const int base = blockIdx.x * MTILE;
  if (base >= Ne) return;

  const int tid  = threadIdx.x;
  const int lane = tid & 31;
  const int wv   = tid >> 5;
  const int hh   = lane >> 4;   // lane half (0/1)
  const int ln   = lane & 15;   // position within half

  if (tid < MTILE) {
    int r = base + tid;
    bool valid = r < Ne;
    stok[tid] = tokIdx[e * B_TOK + (valid ? r : base)];  // replicate row for padding
    sw[tid]   = valid ? tokW[e * B_TOK + r] : 0.f;       // zero weight kills padding
  }
  __syncthreads();

  // gather 32 bf16 token rows into LDS
  {
    const int row = tid >> 3;
    const int c0  = tid & 7;
    const v8bf* src = (const v8bf*)(xb + (size_t)stok[row] * D_DIM);
    #pragma unroll
    for (int c = c0; c < D_DIM / 8; c += 8) xg[row][c] = src[c];
  }
  __syncthreads();

  // ---- phase 1: h = relu(xg @ W1[e] + b1[e]) -> LDS (bf16) ----
  // 128 N-tiles in 32 groups of 4; 8 waves stride the groups; each wave
  // covers both 16-row M-blocks (M=32) so B fragments feed two WMMAs.
  for (int g = wv; g < DFF / 64; g += 8) {
    const int f0 = g * 64;  // columns f0 + j*16 + ln, j = 0..3
    const bf16_t* brow = W1T + ((size_t)e * DFF + f0 + ln) * D_DIM;
    v8f acc[2][4];
    #pragma unroll
    for (int r = 0; r < 2; ++r)
      #pragma unroll
      for (int j = 0; j < 4; ++j)
        acc[r][j] = (v8f){0.f,0.f,0.f,0.f,0.f,0.f,0.f,0.f};
    #pragma unroll 2
    for (int k = 0; k < D_DIM / 32; ++k) {
      const v16bf a0 = cat16(xg[ln][k * 4 + hh],      xg[ln][k * 4 + hh + 2]);
      const v16bf a1 = cat16(xg[16 + ln][k * 4 + hh], xg[16 + ln][k * 4 + hh + 2]);
      const int ko = k * 32 + hh * 8;
      #pragma unroll
      for (int j = 0; j < 4; ++j) {
        const bf16_t* bp = brow + (size_t)j * 16 * D_DIM + ko;
        const v16bf bm = cat16(*(const v8bf*)bp, *(const v8bf*)(bp + 16));
        acc[0][j] = __builtin_amdgcn_wmma_f32_16x16x32_bf16(
            false, a0, false, bm, (short)0, acc[0][j], false, false);
        acc[1][j] = __builtin_amdgcn_wmma_f32_16x16x32_bf16(
            false, a1, false, bm, (short)0, acc[1][j], false, false);
      }
    }
    #pragma unroll
    for (int r = 0; r < 2; ++r) {
      #pragma unroll
      for (int j = 0; j < 4; ++j) {
        const int fcol = f0 + j * 16 + ln;
        const float bias = b1[e * DFF + fcol];
        #pragma unroll
        for (int i = 0; i < 8; ++i) {
          float v = acc[r][j][i] + bias;
          v = v > 0.f ? v : 0.f;
          ((bf16_t*)&hb[r * 16 + hh * 8 + i][0])[fcol] = f2bf(v);
        }
      }
    }
  }
  __syncthreads();

  // ---- phase 2: y = h @ W2[e]; scale by routing weight; atomic add to out ----
  // 32 N-tiles in 8 groups of 4; one group per wave; M=32 per wave.
  for (int g = wv; g < D_DIM / 64; g += 8) {
    const int d0 = g * 64;
    const bf16_t* brow = W2T + ((size_t)e * D_DIM + d0 + ln) * DFF;
    v8f acc[2][4];
    #pragma unroll
    for (int r = 0; r < 2; ++r)
      #pragma unroll
      for (int j = 0; j < 4; ++j)
        acc[r][j] = (v8f){0.f,0.f,0.f,0.f,0.f,0.f,0.f,0.f};
    #pragma unroll 2
    for (int k = 0; k < DFF / 32; ++k) {
      const v16bf a0 = cat16(hb[ln][k * 4 + hh],      hb[ln][k * 4 + hh + 2]);
      const v16bf a1 = cat16(hb[16 + ln][k * 4 + hh], hb[16 + ln][k * 4 + hh + 2]);
      const int ko = k * 32 + hh * 8;
      #pragma unroll
      for (int j = 0; j < 4; ++j) {
        const bf16_t* bp = brow + (size_t)j * 16 * DFF + ko;
        const v16bf bm = cat16(*(const v8bf*)bp, *(const v8bf*)(bp + 16));
        acc[0][j] = __builtin_amdgcn_wmma_f32_16x16x32_bf16(
            false, a0, false, bm, (short)0, acc[0][j], false, false);
        acc[1][j] = __builtin_amdgcn_wmma_f32_16x16x32_bf16(
            false, a1, false, bm, (short)0, acc[1][j], false, false);
      }
    }
    #pragma unroll
    for (int r = 0; r < 2; ++r) {
      #pragma unroll
      for (int j = 0; j < 4; ++j) {
        const int dcol = d0 + j * 16 + ln;
        #pragma unroll
        for (int i = 0; i < 8; ++i) {
          const int m = r * 16 + hh * 8 + i;
          atomicAdd(&out[(size_t)stok[m] * D_DIM + dcol], acc[r][j][i] * sw[m]);
        }
      }
    }
  }
}

// ---------------------------------------------------------------------------
extern "C" void kernel_launch(void* const* d_in, const int* in_sizes, int n_in,
                              void* d_out, int out_size, void* d_ws, size_t ws_size,
                              hipStream_t stream) {
  (void)in_sizes; (void)n_in; (void)out_size; (void)ws_size;
  const float* x  = (const float*)d_in[0];   // [B, D]
  const float* Wg = (const float*)d_in[1];   // [D, E]
  const float* W1 = (const float*)d_in[2];   // [E, D, DFF]
  const float* b1 = (const float*)d_in[3];   // [E, DFF]
  const float* W2 = (const float*)d_in[4];   // [E, DFF, D]
  const float* b2 = (const float*)d_in[5];   // [E, D]
  float* out = (float*)d_out;                // [B, D]

  // workspace carve-up (~51.3 MB total)
  char* ws = (char*)d_ws;
  bf16_t* xb  = (bf16_t*)ws;  ws += (size_t)B_TOK * D_DIM * sizeof(bf16_t);
  bf16_t* W1T = (bf16_t*)ws;  ws += (size_t)E_NUM * D_DIM * DFF * sizeof(bf16_t); // [E,DFF,D]
  bf16_t* W2T = (bf16_t*)ws;  ws += (size_t)E_NUM * D_DIM * DFF * sizeof(bf16_t); // [E,D,DFF]
  int*    cnt = (int*)ws;     ws += 256;                                          // padded
  int* tokIdx = (int*)ws;     ws += (size_t)E_NUM * B_TOK * sizeof(int);
  float* tokW = (float*)ws;   ws += (size_t)E_NUM * B_TOK * sizeof(float);

  hipMemsetAsync(cnt, 0, E_NUM * sizeof(int), stream);

  dim3 tb(32, 8, 1);
  // W1 [E, D, DFF] -> W1T [E, DFF, D]
  transpose_convert_kernel<<<dim3(DFF / 32, D_DIM / 32, E_NUM), tb, 0, stream>>>(
      W1, W1T, D_DIM, DFF);
  // W2 [E, DFF, D] -> W2T [E, D, DFF]
  transpose_convert_kernel<<<dim3(D_DIM / 32, DFF / 32, E_NUM), tb, 0, stream>>>(
      W2, W2T, DFF, D_DIM);

  routing_kernel<<<dim3(B_TOK / 8), dim3(256), 0, stream>>>(
      x, Wg, b2, xb, cnt, tokIdx, tokW, out);

  moe_ffn_kernel<<<dim3(B_TOK / MTILE, E_NUM), dim3(256), 0, stream>>>(
      xb, W1T, b1, W2T, cnt, tokIdx, tokW, out);
}
